// NodeProcessorModule_52510270161469
// MI455X (gfx1250) — matrix operand order repaired and checked by reading, not compiled, not run on hardware
//
#include <hip/hip_runtime.h>
#include <hip/hip_bf16.h>

typedef __attribute__((ext_vector_type(16))) __bf16 v16bf;
typedef __attribute__((ext_vector_type(8)))  __bf16 v8bf;
typedef __attribute__((ext_vector_type(8)))  float  v8f;

#define DLAT 128
#define KIN  384            // 3*DLAT
#define W1_STRIDE 392       // 384 + 8 bf16 pad  (dword stride 196 ≡ 4 mod 64)
#define W23_STRIDE 136      // 128 + 8 bf16 pad  (dword stride  68 ≡ 4 mod 64)
#define H_STRIDE  136       // h-stage row stride (bf16)
#define WAVES_PER_BLOCK 16
#define BLOCK_THREADS (WAVES_PER_BLOCK * 32)
// LDS layout (bytes)
#define LW1_OFF  0
#define LW2_OFF  (DLAT * W1_STRIDE * 2)                    // 100352
#define LW3_OFF  (LW2_OFF + DLAT * W23_STRIDE * 2)         // 135168
#define HB_OFF   (LW3_OFF + DLAT * W23_STRIDE * 2)         // 169984
#define HB_BYTES (16 * H_STRIDE * 2)                       // 4352 per wave
#define SMEM_BYTES (HB_OFF + WAVES_PER_BLOCK * HB_BYTES)   // 239616

__device__ __forceinline__ v8f wmma_bf16(v16bf a, v16bf b, v8f c) {
  return __builtin_amdgcn_wmma_f32_16x16x32_bf16(false, a, false, b, (short)0, c,
                                                 false, false);
}

// ---------------------------------------------------------------- zero agg
__global__ void zero_f4(float4* __restrict__ p, long n4) {
  long i = (long)blockIdx.x * blockDim.x + threadIdx.x;
  if (i < n4) p[i] = make_float4(0.f, 0.f, 0.f, 0.f);
}

// ------------------------------------------------- weights -> bf16, [fo][fi]
__global__ void prep_weights(const float* __restrict__ W1,
                             const float* __restrict__ W2,
                             const float* __restrict__ W3,
                             __bf16* __restrict__ W1t,
                             __bf16* __restrict__ W2t,
                             __bf16* __restrict__ W3t) {
  int i = blockIdx.x * blockDim.x + threadIdx.x;
  if (i < DLAT * KIN) {
    int n = i / KIN, k = i % KIN;
    W1t[i] = (__bf16)W1[k * DLAT + n];          // W1 is [KIN][DLAT] row-major
  }
  if (i < DLAT * DLAT) {
    int n = i >> 7, k = i & (DLAT - 1);
    W2t[i] = (__bf16)W2[k * DLAT + n];
    W3t[i] = (__bf16)W3[k * DLAT + n];
  }
}

// ------------------------------------------------------------- edge scatter
// one wave per edge: 32 lanes x 4 floats = 128; f32 atomics resolve in L2
__global__ void scatter_edges(const float* __restrict__ edge,
                              const int* __restrict__ idx,
                              float* __restrict__ agg, int n_edges) {
  long t = (long)blockIdx.x * blockDim.x + threadIdx.x;
  int e = (int)(t >> 5);
  if (e >= n_edges) return;
  int c = ((int)t & 31) * 4;
  int node = idx[e];
  const float4 v = *(const float4*)(edge + (size_t)e * DLAT + c);
  float* dst = agg + (size_t)node * DLAT + c;
  unsafeAtomicAdd(dst + 0, v.x);
  unsafeAtomicAdd(dst + 1, v.y);
  unsafeAtomicAdd(dst + 2, v.z);
  unsafeAtomicAdd(dst + 3, v.w);
}

// --------------------------------------------- fused 3-layer MLP + LayerNorm
// 16 waves/block, one 16-row node tile per wave; weights staged in LDS once.
__global__ __launch_bounds__(BLOCK_THREADS) void node_mlp(
    const float* __restrict__ node_attr,
    const float* __restrict__ agg_mesh,
    const float* __restrict__ agg_world,
    const __bf16* __restrict__ W1t,
    const __bf16* __restrict__ W2t,
    const __bf16* __restrict__ W3t,
    const float* __restrict__ b1,
    const float* __restrict__ b2,
    const float* __restrict__ b3,
    const float* __restrict__ gamma,
    const float* __restrict__ beta,
    float* __restrict__ out,
    int n_nodes, int n_tiles) {
  extern __shared__ char smem[];
  __bf16* lw1 = (__bf16*)(smem + LW1_OFF);
  __bf16* lw2 = (__bf16*)(smem + LW2_OFF);
  __bf16* lw3 = (__bf16*)(smem + LW3_OFF);

  const int tid  = threadIdx.x;
  const int wave = tid >> 5;
  const int lane = tid & 31;
  const int tile = blockIdx.x * WAVES_PER_BLOCK + wave;
  const bool active = tile < n_tiles;

  // ---- stage weights into LDS (padded rows: conflict-free B-fragment reads)
  for (int i = tid; i < DLAT * 48; i += BLOCK_THREADS) {   // W1: 128 x 384
    int row = i / 48, c = (i % 48) * 8;
    *(uint4*)(lw1 + row * W1_STRIDE + c) = *(const uint4*)(W1t + row * KIN + c);
  }
  for (int i = tid; i < DLAT * 16; i += BLOCK_THREADS) {   // W2,W3: 128 x 128
    int row = i >> 4, c = (i & 15) * 8;
    *(uint4*)(lw2 + row * W23_STRIDE + c) = *(const uint4*)(W2t + row * DLAT + c);
    *(uint4*)(lw3 + row * W23_STRIDE + c) = *(const uint4*)(W3t + row * DLAT + c);
  }
  __syncthreads();
  if (!active) return;

  const int lr   = lane & 15;   // A row within tile / B,D column within tile
  const int half = lane >> 4;   // lane-half per ISA 16-bit WMMA layouts
  int grow = tile * 16 + lr;
  if (grow >= n_nodes) grow = n_nodes - 1;   // clamp: EXEC stays all-ones
  __bf16* hb = (__bf16*)(smem + HB_OFF + wave * HB_BYTES);

  // ---- A fragments for GEMM1: concat(node, agg_mesh, agg_world) on the fly
  // A 16x32 bf16 layout: lane holds row lr; elems 0..15 = K=ks*32+half*16+i
  v16bf a1[12];
#pragma unroll
  for (int ks = 0; ks < 12; ++ks) {
    const float* src;
    int koff;
    if (ks < 4)      { src = node_attr; koff = ks * 32; }
    else if (ks < 8) { src = agg_mesh;  koff = (ks - 4) * 32; }
    else             { src = agg_world; koff = (ks - 8) * 32; }
    const float* p = src + (size_t)grow * DLAT + koff + half * 16;
    float4 x0 = *(const float4*)(p + 0);
    float4 x1 = *(const float4*)(p + 4);
    float4 y0 = *(const float4*)(p + 8);
    float4 y1 = *(const float4*)(p + 12);
    float vals[16] = {x0.x, x0.y, x0.z, x0.w, x1.x, x1.y, x1.z, x1.w,
                      y0.x, y0.y, y0.z, y0.w, y1.x, y1.y, y1.z, y1.w};
#pragma unroll
    for (int i = 0; i < 16; ++i) a1[ks][i] = (__bf16)vals[i];
  }

  // ---- GEMM1: [16x384] x [384x128]; B fragments preloaded from LDS per tile
  v8f h[8];
#pragma unroll
  for (int t = 0; t < 8; ++t) {
    v16bf bfr[12];
#pragma unroll
    for (int ks = 0; ks < 12; ++ks)
      bfr[ks] = *(const v16bf*)(lw1 + (t * 16 + lr) * W1_STRIDE +
                                ks * 32 + half * 16);
    v8f acc = {};
#pragma unroll
    for (int ks = 0; ks < 12; ++ks) acc = wmma_bf16(a1[ks], bfr[ks], acc);
    h[t] = acc;
  }

  // bias + relu, stage bf16 to per-wave LDS in D-layout (lane=col, vgpr=row)
#pragma unroll
  for (int t = 0; t < 8; ++t) {
    float bias = b1[t * 16 + lr];
#pragma unroll
    for (int v = 0; v < 8; ++v) {
      float x = h[t][v] + bias;
      x = x > 0.f ? x : 0.f;
      hb[(half * 8 + v) * H_STRIDE + t * 16 + lr] = (__bf16)x;
    }
  }
  asm volatile("s_wait_dscnt 0x0" ::: "memory");

  // ---- GEMM2
  v16bf a2[4];
#pragma unroll
  for (int ks = 0; ks < 4; ++ks) {
    const __bf16* p = hb + lr * H_STRIDE + ks * 32 + half * 16;
    v8bf lo = *(const v8bf*)(p);
    v8bf hi = *(const v8bf*)(p + 8);
#pragma unroll
    for (int i = 0; i < 8; ++i) { a2[ks][i] = lo[i]; a2[ks][8 + i] = hi[i]; }
  }
#pragma unroll
  for (int t = 0; t < 8; ++t) {
    v16bf bfr[4];
#pragma unroll
    for (int ks = 0; ks < 4; ++ks)
      bfr[ks] = *(const v16bf*)(lw2 + (t * 16 + lr) * W23_STRIDE +
                                ks * 32 + half * 16);
    v8f acc = {};
#pragma unroll
    for (int ks = 0; ks < 4; ++ks) acc = wmma_bf16(a2[ks], bfr[ks], acc);
    h[t] = acc;
  }
  // bias + relu back to LDS (DS ops are in-order within a wave)
#pragma unroll
  for (int t = 0; t < 8; ++t) {
    float bias = b2[t * 16 + lr];
#pragma unroll
    for (int v = 0; v < 8; ++v) {
      float x = h[t][v] + bias;
      x = x > 0.f ? x : 0.f;
      hb[(half * 8 + v) * H_STRIDE + t * 16 + lr] = (__bf16)x;
    }
  }
  asm volatile("s_wait_dscnt 0x0" ::: "memory");

  // ---- GEMM3 (no relu)
  v16bf a3[4];
#pragma unroll
  for (int ks = 0; ks < 4; ++ks) {
    const __bf16* p = hb + lr * H_STRIDE + ks * 32 + half * 16;
    v8bf lo = *(const v8bf*)(p);
    v8bf hi = *(const v8bf*)(p + 8);
#pragma unroll
    for (int i = 0; i < 8; ++i) { a3[ks][i] = lo[i]; a3[ks][8 + i] = hi[i]; }
  }
#pragma unroll
  for (int t = 0; t < 8; ++t) {
    v16bf bfr[4];
#pragma unroll
    for (int ks = 0; ks < 4; ++ks)
      bfr[ks] = *(const v16bf*)(lw3 + (t * 16 + lr) * W23_STRIDE +
                                ks * 32 + half * 16);
    v8f acc = {};
#pragma unroll
    for (int ks = 0; ks < 4; ++ks) acc = wmma_bf16(a3[ks], bfr[ks], acc);
    h[t] = acc;
  }
#pragma unroll
  for (int t = 0; t < 8; ++t) {
    float bias = b3[t * 16 + lr];
#pragma unroll
    for (int v = 0; v < 8; ++v) h[t][v] += bias;
  }

  // ---- LayerNorm over the 128 cols of each row.
  // Row (half*8+v) lives in one 16-lane half: 8 partials/lane + 4 butterflies.
  float mu[8], rs[8];
#pragma unroll
  for (int v = 0; v < 8; ++v) {
    float s = 0.f, ss = 0.f;
#pragma unroll
    for (int t = 0; t < 8; ++t) { float x = h[t][v]; s += x; ss += x * x; }
#pragma unroll
    for (int m = 1; m <= 8; m <<= 1) {   // masks stay inside each 16-lane half
      s  += __shfl_xor(s, m);
      ss += __shfl_xor(ss, m);
    }
    float m0 = s * (1.f / 128.f);
    mu[v] = m0;
    rs[v] = rsqrtf(ss * (1.f / 128.f) - m0 * m0 + 1e-5f);
  }

#pragma unroll
  for (int t = 0; t < 8; ++t) {
    float g  = gamma[t * 16 + lr];
    float be = beta[t * 16 + lr];
#pragma unroll
    for (int v = 0; v < 8; ++v) {
      int row = tile * 16 + half * 8 + v;
      if (row < n_nodes)
        out[(size_t)row * DLAT + t * 16 + lr] =
            (h[t][v] - mu[v]) * rs[v] * g + be;
    }
  }
}

extern "C" void kernel_launch(void* const* d_in, const int* in_sizes, int n_in,
                              void* d_out, int out_size, void* d_ws,
                              size_t ws_size, hipStream_t stream) {
  const float* node_attr  = (const float*)d_in[0];
  const float* edge_attr  = (const float*)d_in[1];
  const float* edge_world = (const float*)d_in[2];
  const int*   mesh_idx   = (const int*)d_in[3];
  const int*   world_idx  = (const int*)d_in[4];
  const float* W1 = (const float*)d_in[5];
  const float* b1 = (const float*)d_in[6];
  const float* W2 = (const float*)d_in[7];
  const float* b2 = (const float*)d_in[8];
  const float* W3 = (const float*)d_in[9];
  const float* b3 = (const float*)d_in[10];
  const float* ln_g = (const float*)d_in[11];
  const float* ln_b = (const float*)d_in[12];

  const int n_nodes = in_sizes[0] / DLAT;
  const int n_mesh  = in_sizes[1] / DLAT;
  const int n_world = in_sizes[2] / DLAT;

  // workspace layout: agg_mesh | agg_world | W1t | W2t | W3t
  float*  agg_mesh  = (float*)d_ws;
  float*  agg_world = agg_mesh + (size_t)n_nodes * DLAT;
  __bf16* W1t = (__bf16*)(agg_world + (size_t)n_nodes * DLAT);
  __bf16* W2t = W1t + DLAT * KIN;
  __bf16* W3t = W2t + DLAT * DLAT;

  long n4 = (2L * n_nodes * DLAT) / 4;
  zero_f4<<<(int)((n4 + 255) / 256), 256, 0, stream>>>((float4*)agg_mesh, n4);

  prep_weights<<<(DLAT * KIN + 255) / 256, 256, 0, stream>>>(W1, W2, W3, W1t,
                                                             W2t, W3t);

  long tm = (long)n_mesh * 32;
  scatter_edges<<<(int)((tm + 255) / 256), 256, 0, stream>>>(edge_attr,
                                                             mesh_idx,
                                                             agg_mesh, n_mesh);
  long tw = (long)n_world * 32;
  scatter_edges<<<(int)((tw + 255) / 256), 256, 0, stream>>>(edge_world,
                                                             world_idx,
                                                             agg_world,
                                                             n_world);

  int n_tiles  = (n_nodes + 15) / 16;
  int n_blocks = (n_tiles + WAVES_PER_BLOCK - 1) / WAVES_PER_BLOCK;
  static bool attr_set = false;
  if (!attr_set) {
    hipFuncSetAttribute((const void*)node_mlp,
                        hipFuncAttributeMaxDynamicSharedMemorySize, SMEM_BYTES);
    attr_set = true;
  }
  node_mlp<<<n_blocks, BLOCK_THREADS, SMEM_BYTES, stream>>>(
      node_attr, agg_mesh, agg_world, W1t, W2t, W3t, b1, b2, b3, ln_g, ln_b,
      (float*)d_out, n_nodes, n_tiles);
}
